// NKMLPPredictor_29678224016204
// MI455X (gfx1250) — matrix operand order repaired and checked by reading, not compiled, not run on hardware
//
#include <hip/hip_runtime.h>
#include <hip/hip_bf16.h>

// ---------------- problem constants (from reference) ----------------
#define F_DIM   128           // node feature dim
#define D_IN    268           // 2*F + 12
#define KPAD    288           // D_IN padded to 9 * 32 (WMMA K-steps)
#define C1      28            // layer-1 out channels
#define NPAD    32            // C1 padded to 2 WMMA n-tiles
#define C2      8             // layer-2 out channels
#define WAVES   4             // waves per block in layer-1 kernel
#define TPB1    (WAVES * 32)  // 128 threads

typedef __attribute__((ext_vector_type(16))) _Float16 v16h;
typedef __attribute__((ext_vector_type(8)))  float    v8f;

// -------------------------------------------------------------------
// Prep: W1 (f32 [28][268]) -> f16 padded [NPAD][KPAD], zeros in pads.
// -------------------------------------------------------------------
__global__ void k_prep_w1(const float* __restrict__ W1, _Float16* __restrict__ w1h) {
    int i = blockIdx.x * blockDim.x + threadIdx.x;
    if (i < NPAD * KPAD) {
        int n = i / KPAD, k = i % KPAD;
        float v = (n < C1 && k < D_IN) ? W1[n * D_IN + k] : 0.0f;
        w1h[i] = (_Float16)v;
    }
}

// -------------------------------------------------------------------
// Layer 1: gather + concat + (x @ W1^T) via WMMA f16->f32.
// Each wave: 16 edges x 32 (padded) channels, K = 288 in 9 steps.
// Emits per-channel sum / sumsq for BN stats (bias cancels in BN).
// y1 stored as f16 [Epad][NPAD].
// -------------------------------------------------------------------
__global__ void __launch_bounds__(TPB1)
k_layer1(const float* __restrict__ h,
         const float* __restrict__ ef,
         const int*   __restrict__ src,
         const int*   __restrict__ dst,
         const _Float16* __restrict__ w1h,
         _Float16* __restrict__ y1,
         float* __restrict__ stat,     // [0..31]=sum, [32..63]=sumsq
         int E)
{
    __shared__ _Float16 BW[NPAD * KPAD];          // 18 KB  W1 f16 (n-major)
    __shared__ _Float16 XA[WAVES][16 * KPAD];     // 9 KB/wave  A-tiles
    __shared__ _Float16 YT[WAVES][16 * NPAD];     // 1 KB/wave  out bounce
    __shared__ float ssum[NPAD], ssq[NPAD];

    const int tid  = threadIdx.x;
    const int wave = tid >> 5;
    const int lane = tid & 31;

    // cooperative load of padded W1 into LDS (1152 x uint4)
    {
        const uint4* g = (const uint4*)w1h;
        uint4* s = (uint4*)BW;
        for (int i = tid; i < (NPAD * KPAD * 2) / 16; i += TPB1) s[i] = g[i];
    }
    if (tid < NPAD) { ssum[tid] = 0.0f; ssq[tid] = 0.0f; }

    const long tileBase = (long)(blockIdx.x * WAVES + wave) * 16;
    _Float16* xa = &XA[wave][0];

    // zero K-pad columns [268..288)
    for (int i = lane; i < 16 * (KPAD - D_IN); i += 32) {
        int r = i / (KPAD - D_IN), c = D_IN + i % (KPAD - D_IN);
        xa[r * KPAD + c] = (_Float16)0.0f;
    }
    // gather + f16 convert: 16 rows x 67 float4 (32 h_u + 32 h_v + 3 ef)
    for (int i = lane; i < 16 * 67; i += 32) {
        int r = i / 67, p = i % 67;
        long e = tileBase + r;
        float4 v = make_float4(0.f, 0.f, 0.f, 0.f);
        if (e < E) {
            if (p < 32) {
                int s0 = src[e];
                v = ((const float4*)(h + (long)s0 * F_DIM))[p];
            } else if (p < 64) {
                int d0 = dst[e];
                v = ((const float4*)(h + (long)d0 * F_DIM))[p - 32];
            } else {
                v = ((const float4*)(ef + e * 12))[p - 64];
            }
        }
        int c = (p < 32) ? p * 4 : (p < 64) ? 128 + (p - 32) * 4 : 256 + (p - 64) * 4;
        _Float16* d = xa + r * KPAD + c;
        d[0] = (_Float16)v.x; d[1] = (_Float16)v.y;
        d[2] = (_Float16)v.z; d[3] = (_Float16)v.w;
    }
    __syncthreads();

    // --- WMMA main loop: D[16x16] += A[16x32] * B[32x16], 9 k-steps, 2 n-tiles
    v8f c0 = {}; v8f c1 = {};
    const int arow = lane & 15;
    const int ak0  = (lane < 16) ? 0 : 8;   // A: lanes<16 hold K{0..7,16..23}
    const int bn0  = lane & 15;
    const int bk0  = (lane < 16) ? 0 : 8;   // B: lanes<16 hold K{0..7,16..23}
#pragma unroll
    for (int t = 0; t < KPAD / 32; ++t) {
        const int kb = t * 32;
        union { v16h v; uint4 u[2]; } A, B0, B1;
        A.u[0]  = *(const uint4*)(xa + arow * KPAD + kb + ak0);
        A.u[1]  = *(const uint4*)(xa + arow * KPAD + kb + ak0 + 16);
        B0.u[0] = *(const uint4*)(BW + bn0 * KPAD + kb + bk0);
        B0.u[1] = *(const uint4*)(BW + bn0 * KPAD + kb + bk0 + 16);
        B1.u[0] = *(const uint4*)(BW + (bn0 + 16) * KPAD + kb + bk0);
        B1.u[1] = *(const uint4*)(BW + (bn0 + 16) * KPAD + kb + bk0 + 16);
        c0 = __builtin_amdgcn_wmma_f32_16x16x32_f16(false, A.v, false, B0.v,
                                                    (short)0, c0, false, false);
        c1 = __builtin_amdgcn_wmma_f32_16x16x32_f16(false, A.v, false, B1.v,
                                                    (short)0, c1, false, false);
    }

    // --- BN stats: lane holds column bn0 (c0) / bn0+16 (c1), 8 rows each.
    // Rows beyond E were zero-filled -> contribute 0. Bias omitted (cancels).
    {
        float s0 = 0.f, q0 = 0.f, s1 = 0.f, q1 = 0.f;
#pragma unroll
        for (int r = 0; r < 8; ++r) {
            float a = c0[r]; s0 += a; q0 += a * a;
            float b = c1[r]; s1 += b; q1 += b * b;
        }
        atomicAdd(&ssum[bn0], s0);
        atomicAdd(&ssq[bn0],  q0);
        if (bn0 + 16 < C1) {
            atomicAdd(&ssum[bn0 + 16], s1);
            atomicAdd(&ssq[bn0 + 16],  q1);
        }
    }

    // --- bounce D through LDS (f16) then coalesced b128 stores
    {
        _Float16* yt = &YT[wave][0];
        const int m0 = (lane < 16) ? 0 : 8;
#pragma unroll
        for (int r = 0; r < 8; ++r) {
            yt[(m0 + r) * NPAD + bn0]      = (_Float16)c0[r];
            yt[(m0 + r) * NPAD + bn0 + 16] = (_Float16)c1[r];
        }
    }
    __syncthreads();
    {
        const int er = lane & 15, chunk = lane >> 4;   // 2 x 32B per row
        const long ge = tileBase + er;
        const uint4* s = (const uint4*)(&YT[wave][0] + er * NPAD + chunk * 16);
        uint4* g = (uint4*)(y1 + ge * NPAD + chunk * 16);
        g[0] = s[0]; g[1] = s[1];
    }
    __syncthreads();
    if (tid < NPAD) {
        atomicAdd(&stat[tid],        ssum[tid]);
        atomicAdd(&stat[NPAD + tid], ssq[tid]);
    }
}

// -------------------------------------------------------------------
// BN affine from global stats: a = gamma*rsqrt(var+eps), c = beta - mean*a
// stat layout: sum[stride], sumsq[stride]; aff layout: a[stride], c[stride]
// -------------------------------------------------------------------
__global__ void k_affine(const float* __restrict__ stat,
                         const float* __restrict__ gamma,
                         const float* __restrict__ beta,
                         float* __restrict__ aff,
                         int C, int stride, float invE)
{
    int n = threadIdx.x;
    if (n < C) {
        float mean = stat[n] * invE;
        float var  = stat[stride + n] * invE - mean * mean;
        float a = gamma[n] * rsqrtf(var + 1e-5f);
        aff[n] = a;
        aff[stride + n] = beta[n] - mean * a;
    } else if (n < stride) {
        aff[n] = 0.0f; aff[stride + n] = 0.0f;
    }
}

// -------------------------------------------------------------------
// Layer 2: affine1+ReLU on y1, tiny 28->8 GEMM on VALU, BN2 stats.
// -------------------------------------------------------------------
__global__ void k_layer2(const _Float16* __restrict__ y1,   // [Epad][NPAD]
                         const float* __restrict__ aff1,    // a[32], c[32]
                         const float* __restrict__ W2,      // [8][28]
                         _Float16* __restrict__ y2,         // [Epad][8]
                         float* __restrict__ stat2,         // sum[8], sq[8]
                         int E)
{
    __shared__ float sW[C2 * C1];
    __shared__ float sA[C1], sC[C1];
    __shared__ float s2[C2], q2[C2];
    const int tid = threadIdx.x;
    for (int i = tid; i < C2 * C1; i += blockDim.x) sW[i] = W2[i];
    if (tid < C1) { sA[tid] = aff1[tid]; sC[tid] = aff1[NPAD + tid]; }
    if (tid < C2) { s2[tid] = 0.0f; q2[tid] = 0.0f; }
    __syncthreads();

    const long e = (long)blockIdx.x * blockDim.x + tid;
    float out[C2];
#pragma unroll
    for (int o = 0; o < C2; ++o) out[o] = 0.0f;

    if (e < E) {
        union { uint4 u[4]; _Float16 hh[NPAD]; } X;
        const uint4* g = (const uint4*)(y1 + e * NPAD);
        X.u[0] = g[0]; X.u[1] = g[1]; X.u[2] = g[2]; X.u[3] = g[3];
        float s1v[C1];
#pragma unroll
        for (int k = 0; k < C1; ++k) {
            float v = (float)X.hh[k] * sA[k] + sC[k];
            s1v[k] = v > 0.0f ? v : 0.0f;
        }
#pragma unroll
        for (int o = 0; o < C2; ++o) {
            float acc = 0.0f;
#pragma unroll
            for (int k = 0; k < C1; ++k) acc += sW[o * C1 + k] * s1v[k];
            out[o] = acc;                       // bias b2 omitted: cancels in BN
        }
        union { uint4 u; _Float16 hh[C2]; } Y;
#pragma unroll
        for (int o = 0; o < C2; ++o) Y.hh[o] = (_Float16)out[o];
        *(uint4*)(y2 + e * C2) = Y.u;
    }
    // wave-level reduce, then LDS, then global atomics
#pragma unroll
    for (int o = 0; o < C2; ++o) {
        float sv = out[o], qv = out[o] * out[o];
        for (int m = 16; m >= 1; m >>= 1) {
            sv += __shfl_xor(sv, m);
            qv += __shfl_xor(qv, m);
        }
        if ((tid & 31) == 0) { atomicAdd(&s2[o], sv); atomicAdd(&q2[o], qv); }
    }
    __syncthreads();
    if (tid < C2) {
        atomicAdd(&stat2[tid],      s2[tid]);
        atomicAdd(&stat2[C2 + tid], q2[tid]);
    }
}

// -------------------------------------------------------------------
// Layer 3: affine2+ReLU on y2, dot with W3, + b3 -> score[E]
// -------------------------------------------------------------------
__global__ void k_layer3(const _Float16* __restrict__ y2,
                         const float* __restrict__ aff2,  // a[8], c[8]
                         const float* __restrict__ W3,    // [1][8]
                         const float* __restrict__ b3,    // [1]
                         float* __restrict__ out, int E)
{
    __shared__ float sA[C2], sC[C2], sW[C2];
    __shared__ float sb;
    const int tid = threadIdx.x;
    if (tid < C2) { sA[tid] = aff2[tid]; sC[tid] = aff2[C2 + tid]; sW[tid] = W3[tid]; }
    if (tid == 0) sb = b3[0];
    __syncthreads();
    const long e = (long)blockIdx.x * blockDim.x + tid;
    if (e < E) {
        union { uint4 u; _Float16 hh[C2]; } X;
        X.u = *(const uint4*)(y2 + e * C2);
        float acc = sb;
#pragma unroll
        for (int o = 0; o < C2; ++o) {
            float v = (float)X.hh[o] * sA[o] + sC[o];
            v = v > 0.0f ? v : 0.0f;
            acc += sW[o] * v;
        }
        out[e] = acc;
    }
}

// -------------------------------------------------------------------
extern "C" void kernel_launch(void* const* d_in, const int* in_sizes, int n_in,
                              void* d_out, int out_size, void* d_ws, size_t ws_size,
                              hipStream_t stream) {
    const float* h      = (const float*)d_in[0];
    const float* ef     = (const float*)d_in[1];
    const int*   src    = (const int*)d_in[2];
    const int*   dst    = (const int*)d_in[3];
    const float* W1     = (const float*)d_in[4];
    // b1 = d_in[5]  : cancels in BN
    const float* gamma1 = (const float*)d_in[6];
    const float* beta1  = (const float*)d_in[7];
    const float* W2     = (const float*)d_in[8];
    // b2 = d_in[9]  : cancels in BN
    const float* gamma2 = (const float*)d_in[10];
    const float* beta2  = (const float*)d_in[11];
    const float* W3     = (const float*)d_in[12];
    const float* b3     = (const float*)d_in[13];

    const int E    = in_sizes[2];
    const int Epad = ((E + WAVES * 16 - 1) / (WAVES * 16)) * (WAVES * 16);

    // workspace layout
    float* stat = (float*)d_ws;                         // sum1[32] sq1[32] sum2[8] sq2[8]
    float* aff1 = stat + 128;                           // a[32], c[32]
    float* aff2 = stat + 192;                           // a[8], c[8]
    _Float16* w1h = (_Float16*)((char*)d_ws + 1024);    // 18432 B
    _Float16* y1  = (_Float16*)((char*)d_ws + 32768);   // Epad*32 f16
    _Float16* y2  = (_Float16*)((char*)d_ws + 32768 + (size_t)Epad * NPAD * 2);

    hipMemsetAsync(d_ws, 0, 1024, stream);   // zero stats each call (graph-safe)

    k_prep_w1<<<(NPAD * KPAD + 255) / 256, 256, 0, stream>>>(W1, w1h);
    k_layer1<<<Epad / (WAVES * 16), TPB1, 0, stream>>>(h, ef, src, dst, w1h, y1, stat, E);
    k_affine<<<1, 32, 0, stream>>>(stat, gamma1, beta1, aff1, C1, NPAD, 1.0f / (float)E);
    k_layer2<<<(E + 255) / 256, 256, 0, stream>>>(y1, aff1, W2, y2, stat + 64, E);
    k_affine<<<1, 32, 0, stream>>>(stat + 64, gamma2, beta2, aff2, C2, C2, 1.0f / (float)E);
    k_layer3<<<(E + 255) / 256, 256, 0, stream>>>(y2, aff2, W3, b3, (float*)d_out, E);
}
